// DynamicGraphConvolution_81123342287014
// MI455X (gfx1250) — compile-verified, matrix operand order checked by reading, not run on hardware
//
#include <hip/hip_runtime.h>
#include <math.h>

// ---------------- problem constants ----------------
static constexpr int kB = 32;
static constexpr int kC = 64;
static constexpr int kN = 1024;

// ---------------- WMMA types ----------------
typedef __attribute__((ext_vector_type(16))) __bf16 v16bf;
typedef __attribute__((ext_vector_type(8)))  __bf16 v8bf;
typedef __attribute__((ext_vector_type(8)))  float  v8f;

// Fragment loader for 16x32 bf16 tiles (A row-major with ld, or Bᵀ with ld).
// ISA layout: lanes 0-15 hold row r, K {k0..k0+7, k0+16..23}; lanes 16-31 hold
// row r, K {k0+8..15, k0+24..31}.  base must already include the k0 offset.
__device__ __forceinline__ v16bf load_frag(const __bf16* base, int ld, int lane) {
    const int r = lane & 15;
    const int h = (lane >> 4) & 1;
    const __bf16* p = base + r * ld + h * 8;
    v8bf lo = *(const v8bf*)(p);
    v8bf hi = *(const v8bf*)(p + 16);
    v16bf f;
#pragma unroll
    for (int i = 0; i < 8; ++i) { f[i] = lo[i]; f[i + 8] = hi[i]; }
    return f;
}

__device__ __forceinline__ v8f wmma_bf16(v16bf a, v16bf b, v8f c) {
    return __builtin_amdgcn_wmma_f32_16x16x32_bf16(
        /*neg_a=*/false, a, /*neg_b=*/false, b,
        /*c_mod=*/(short)0, c, /*reuse_a=*/false, /*reuse_b=*/false);
}

__device__ __forceinline__ float lrelu(float v) { return v >= 0.f ? v : 0.2f * v; }

// ---------------- precision conversion kernels ----------------
__global__ void dgc_cvt_bf16(const float* __restrict__ in, __bf16* __restrict__ out, int n) {
    for (int i = blockIdx.x * blockDim.x + threadIdx.x; i < n; i += gridDim.x * blockDim.x)
        out[i] = (__bf16)in[i];
}

// xT[b][n][c] = x[b][c][n]  (bf16)
__global__ void dgc_xpose_cvt(const float* __restrict__ x, __bf16* __restrict__ xT) {
    const int total = kB * kN * kC;
    for (int i = blockIdx.x * blockDim.x + threadIdx.x; i < total; i += gridDim.x * blockDim.x) {
        int b = i / (kN * kC);
        int r = i - b * (kN * kC);
        int n = r / kC;
        int c = r - n * kC;
        xT[i] = (__bf16)x[(b * kC + c) * kN + n];
    }
}

// ---------------- K1: t = lrelu(X @ W_saT), stored transposed tT[b][m][c] ----
__global__ __launch_bounds__(256)
void dgc_k1_static_adj(const __bf16* __restrict__ xb, const __bf16* __restrict__ Wsab,
                       __bf16* __restrict__ tT) {
    const int lane = threadIdx.x & 31;
    const int tile = blockIdx.x * 8 + (threadIdx.x >> 5);
    const int numTiles = kB * (kC / 16) * (kN / 16);
    if (tile >= numTiles) return;
    const int b  = tile / ((kC / 16) * (kN / 16));
    const int rm = tile - b * ((kC / 16) * (kN / 16));
    const int c0 = (rm / (kN / 16)) * 16;
    const int m0 = (rm % (kN / 16)) * 16;

    v8f acc = {};
    for (int k0 = 0; k0 < kN; k0 += 32) {
        v16bf a  = load_frag(xb + (b * kC + c0) * kN + k0, kN, lane);     // A: X rows c
        v16bf bf = load_frag(Wsab + m0 * kN + k0, kN, lane);              // BT: W_sa rows m
        acc = wmma_bf16(a, bf, acc);
    }
    const int col = m0 + (lane & 15);
    const int h8  = ((lane >> 4) & 1) * 8;
#pragma unroll
    for (int v = 0; v < 8; ++v) {
        int row = c0 + v + h8;
        tT[(b * kN + col) * kC + row] = (__bf16)lrelu(acc[v]);
    }
}

// -------- K2: x1 = x + lrelu(W_sw @ t + b_sw), stored row-major bf16 --------
__global__ __launch_bounds__(256)
void dgc_k2_static_weight(const __bf16* __restrict__ Wswb, const __bf16* __restrict__ tT,
                          const float* __restrict__ x, const float* __restrict__ b_sw,
                          __bf16* __restrict__ x1b) {
    const int lane = threadIdx.x & 31;
    const int tile = blockIdx.x * 8 + (threadIdx.x >> 5);
    const int numTiles = kB * (kC / 16) * (kN / 16);
    if (tile >= numTiles) return;
    const int b  = tile / ((kC / 16) * (kN / 16));
    const int rm = tile - b * ((kC / 16) * (kN / 16));
    const int c0 = (rm / (kN / 16)) * 16;
    const int n0 = (rm % (kN / 16)) * 16;

    v8f acc = {};
#pragma unroll
    for (int k0 = 0; k0 < kC; k0 += 32) {
        v16bf a  = load_frag(Wswb + c0 * kC + k0, kC, lane);                 // A: W_sw rows o
        v16bf bf = load_frag(tT + (b * kN + n0) * kC + k0, kC, lane);        // BT: tT rows n
        acc = wmma_bf16(a, bf, acc);
    }
    const int col = n0 + (lane & 15);
    const int h8  = ((lane >> 4) & 1) * 8;
#pragma unroll
    for (int v = 0; v < 8; ++v) {
        int row = c0 + v + h8;
        float val = x[(b * kC + row) * kN + col] + lrelu(acc[v] + b_sw[row]);
        x1b[(b * kC + row) * kN + col] = (__bf16)val;
    }
}

// ---------------- K3: energy = X @ XT  (fp32 out) ----------------
__global__ __launch_bounds__(256)
void dgc_k3_energy(const __bf16* __restrict__ xb, float* __restrict__ energy) {
    const int lane = threadIdx.x & 31;
    const int tile = blockIdx.x * 8 + (threadIdx.x >> 5);
    const int numTiles = kB * (kC / 16) * (kC / 16);
    if (tile >= numTiles) return;
    const int b  = tile >> 4;
    const int rm = tile & 15;
    const int c0 = (rm >> 2) * 16;
    const int d0 = (rm & 3) * 16;

    v8f acc = {};
    for (int k0 = 0; k0 < kN; k0 += 32) {
        v16bf a  = load_frag(xb + (b * kC + c0) * kN + k0, kN, lane);     // A: X rows c
        v16bf bf = load_frag(xb + (b * kC + d0) * kN + k0, kN, lane);     // BT: X rows d
        acc = wmma_bf16(a, bf, acc);
    }
    const int col = d0 + (lane & 15);
    const int h8  = ((lane >> 4) & 1) * 8;
#pragma unroll
    for (int v = 0; v < 8; ++v) {
        int row = c0 + v + h8;
        energy[(b * kC + row) * kC + col] = acc[v];
    }
}

// ---------------- K4a: energy_new = rowmax(energy) - energy ----------------
__global__ __launch_bounds__(64)
void dgc_k4a_rowmax(const float* __restrict__ energy, float* __restrict__ en) {
    const int bc = blockIdx.x;           // b*C + c
    __shared__ float sm[64];
    float v = energy[bc * kC + threadIdx.x];
    sm[threadIdx.x] = v;
    __syncthreads();
    for (int st = 32; st > 0; st >>= 1) {
        if (threadIdx.x < st) sm[threadIdx.x] = fmaxf(sm[threadIdx.x], sm[threadIdx.x + st]);
        __syncthreads();
    }
    en[bc * kC + threadIdx.x] = sm[0] - v;
}

// ---- K4b: per-channel BN over (B, d) -> attention (bf16, row-major) --------
__global__ __launch_bounds__(256)
void dgc_k4b_bn_attn(const float* __restrict__ en, const float* __restrict__ bn_g,
                     const float* __restrict__ bn_b, __bf16* __restrict__ attn) {
    const int c = blockIdx.x;
    __shared__ float ssum[256], ssq[256];
    float s = 0.f, q = 0.f;
    for (int t = threadIdx.x; t < kB * kC; t += 256) {
        int b = t >> 6, d = t & 63;
        float v = en[(b * kC + c) * kC + d];
        s += v; q += v * v;
    }
    ssum[threadIdx.x] = s; ssq[threadIdx.x] = q;
    __syncthreads();
    for (int st = 128; st > 0; st >>= 1) {
        if (threadIdx.x < st) { ssum[threadIdx.x] += ssum[threadIdx.x + st];
                                ssq[threadIdx.x]  += ssq[threadIdx.x + st]; }
        __syncthreads();
    }
    const float inv = 1.f / (float)(kB * kC);
    const float mu  = ssum[0] * inv;
    const float var = ssq[0] * inv - mu * mu;
    const float rs  = rsqrtf(var + 1e-5f);
    const float g = bn_g[c], be = bn_b[c];
    for (int t = threadIdx.x; t < kB * kC; t += 256) {
        int b = t >> 6, d = t & 63;
        int idx = (b * kC + c) * kC + d;
        attn[idx] = (__bf16)(g * (en[idx] - mu) * rs + be);
    }
}

// -------- K5: x_glb = gamma*(attn @ X) + x, stored transposed xgT[b][n][c] --
__global__ __launch_bounds__(256)
void dgc_k5_xglb(const __bf16* __restrict__ attn, const __bf16* __restrict__ xT,
                 const float* __restrict__ x, const float* __restrict__ gammap,
                 __bf16* __restrict__ xgT) {
    const int lane = threadIdx.x & 31;
    const int tile = blockIdx.x * 8 + (threadIdx.x >> 5);
    const int numTiles = kB * (kC / 16) * (kN / 16);
    if (tile >= numTiles) return;
    const int b  = tile / ((kC / 16) * (kN / 16));
    const int rm = tile - b * ((kC / 16) * (kN / 16));
    const int c0 = (rm / (kN / 16)) * 16;
    const int n0 = (rm % (kN / 16)) * 16;
    const float gamma = gammap[0];

    v8f acc = {};
#pragma unroll
    for (int k0 = 0; k0 < kC; k0 += 32) {
        v16bf a  = load_frag(attn + b * kC * kC + c0 * kC + k0, kC, lane);   // A: attn rows c
        v16bf bf = load_frag(xT + (b * kN + n0) * kC + k0, kC, lane);        // BT: xT rows n
        acc = wmma_bf16(a, bf, acc);
    }
    const int col = n0 + (lane & 15);
    const int h8  = ((lane >> 4) & 1) * 8;
#pragma unroll
    for (int v = 0; v < 8; ++v) {
        int row = c0 + v + h8;
        float val = gamma * acc[v] + x[(b * kC + row) * kN + col];
        xgT[(b * kN + col) * kC + row] = (__bf16)val;
    }
}

// ---- K6: fused dyn_adj -> y -> W_dw.  dyn_adj never touches HBM. ----------
// block: 512 threads = 16 waves, computes a [C=64, Mt=64] output slab.
__global__ __launch_bounds__(512)
void dgc_k6_dynamic(const __bf16* __restrict__ Wcmb, const __bf16* __restrict__ xgT,
                    const __bf16* __restrict__ x1b, const __bf16* __restrict__ Wdwb,
                    const float* __restrict__ b_cm, const float* __restrict__ b_dw,
                    float* __restrict__ y) {
    __shared__ __bf16 Plds[64 * 64];   // dyn_adj chunk, [m][n] layout
    __shared__ __bf16 Ylds[64 * 64];   // intermediate y,  [m][c] layout

    const int lane = threadIdx.x & 31;
    const int wave = threadIdx.x >> 5;   // 0..15
    const int wi = wave >> 2;            // row-tile   (0..3)
    const int wj = wave & 3;             // col-tile   (0..3)
    const int b   = blockIdx.x >> 4;     // batch
    const int m0g = (blockIdx.x & 15) * 64;

    const int l15 = lane & 15;
    const int h8  = ((lane >> 4) & 1) * 8;

    v8f yacc = {};
    for (int nc = 0; nc < kN / 64; ++nc) {
        const int n0 = nc * 64;
        // ---- phase 1: P = sigmoid(W_cm[n0:n0+64,:] @ Xg[:, m-tile] + b_cm) ----
        v8f p = {};
#pragma unroll
        for (int k0 = 0; k0 < kC; k0 += 32) {
            v16bf a  = load_frag(Wcmb + (n0 + wi * 16) * kC + k0, kC, lane);
            v16bf bf = load_frag(xgT + (b * kN + m0g + wj * 16) * kC + k0, kC, lane);
            p = wmma_bf16(a, bf, p);
        }
#pragma unroll
        for (int v = 0; v < 8; ++v) {
            int nrow = wi * 16 + v + h8;             // n within chunk
            int mcol = wj * 16 + l15;                // m within tile
            float val = p[v] + b_cm[n0 + nrow];
            float sg  = 1.f / (1.f + __expf(-val));
            Plds[mcol * 64 + nrow] = (__bf16)sg;
        }
        __syncthreads();
        // ---- phase 2: Y += X1[:, n-chunk] @ P ----
#pragma unroll
        for (int k0 = 0; k0 < 64; k0 += 32) {
            v16bf a  = load_frag(x1b + (b * kC + wi * 16) * kN + n0 + k0, kN, lane);
            v16bf bf = load_frag(Plds + (wj * 16) * 64 + k0, 64, lane);
            yacc = wmma_bf16(a, bf, yacc);
        }
        __syncthreads();   // before Plds is overwritten next iteration
    }

    // ---- stage Y (after lrelu) into LDS as Bᵀ for the W_dw matmul ----
#pragma unroll
    for (int v = 0; v < 8; ++v) {
        int crow = wi * 16 + v + h8;
        int mcol = wj * 16 + l15;
        Ylds[mcol * 64 + crow] = (__bf16)lrelu(yacc[v]);
    }
    __syncthreads();

    // ---- phase 3: out = lrelu(W_dw @ Y + b_dw) ----
    v8f o = {};
#pragma unroll
    for (int k0 = 0; k0 < kC; k0 += 32) {
        v16bf a  = load_frag(Wdwb + (wi * 16) * kC + k0, kC, lane);
        v16bf bf = load_frag(Ylds + (wj * 16) * 64 + k0, 64, lane);
        o = wmma_bf16(a, bf, o);
    }
#pragma unroll
    for (int v = 0; v < 8; ++v) {
        int row = wi * 16 + v + h8;
        int col = m0g + wj * 16 + l15;
        y[(b * kC + row) * kN + col] = lrelu(o[v] + b_dw[row]);
    }
}

// ---------------- host side ----------------
extern "C" void kernel_launch(void* const* d_in, const int* in_sizes, int n_in,
                              void* d_out, int out_size, void* d_ws, size_t ws_size,
                              hipStream_t stream) {
    (void)in_sizes; (void)n_in; (void)out_size; (void)ws_size;
    const float* x     = (const float*)d_in[0];
    const float* W_sa  = (const float*)d_in[1];
    const float* W_sw  = (const float*)d_in[2];
    const float* b_sw  = (const float*)d_in[3];
    const float* bn_g  = (const float*)d_in[4];
    const float* bn_b  = (const float*)d_in[5];
    const float* gamma = (const float*)d_in[6];
    const float* W_cm  = (const float*)d_in[7];
    const float* b_cm  = (const float*)d_in[8];
    const float* W_dw  = (const float*)d_in[9];
    const float* b_dw  = (const float*)d_in[10];
    float* y = (float*)d_out;

    char* ws = (char*)d_ws;
    size_t off = 0;
    auto alloc = [&](size_t bytes) -> void* {
        void* p = ws + off;
        off = (off + bytes + 255) & ~(size_t)255;
        return p;
    };
    __bf16* xb    = (__bf16*)alloc((size_t)kB * kC * kN * 2);
    __bf16* xT    = (__bf16*)alloc((size_t)kB * kN * kC * 2);
    __bf16* Wsab  = (__bf16*)alloc((size_t)kN * kN * 2);
    __bf16* Wswb  = (__bf16*)alloc((size_t)kC * kC * 2);
    __bf16* Wcmb  = (__bf16*)alloc((size_t)kN * kC * 2);
    __bf16* Wdwb  = (__bf16*)alloc((size_t)kC * kC * 2);
    __bf16* tT    = (__bf16*)alloc((size_t)kB * kN * kC * 2);
    __bf16* x1b   = (__bf16*)alloc((size_t)kB * kC * kN * 2);
    float*  energ = (float*) alloc((size_t)kB * kC * kC * 4);
    float*  en    = (float*) alloc((size_t)kB * kC * kC * 4);
    __bf16* attn  = (__bf16*)alloc((size_t)kB * kC * kC * 2);
    __bf16* xgT   = (__bf16*)alloc((size_t)kB * kN * kC * 2);

    auto cvtGrid = [](int n) { int g = (n + 255) / 256; return g > 4096 ? 4096 : g; };

    // precision conversion
    dgc_cvt_bf16<<<cvtGrid(kB * kC * kN), 256, 0, stream>>>(x, xb, kB * kC * kN);
    dgc_xpose_cvt<<<cvtGrid(kB * kN * kC), 256, 0, stream>>>(x, xT);
    dgc_cvt_bf16<<<cvtGrid(kN * kN), 256, 0, stream>>>(W_sa, Wsab, kN * kN);
    dgc_cvt_bf16<<<cvtGrid(kC * kC), 256, 0, stream>>>(W_sw, Wswb, kC * kC);
    dgc_cvt_bf16<<<cvtGrid(kN * kC), 256, 0, stream>>>(W_cm, Wcmb, kN * kC);
    dgc_cvt_bf16<<<cvtGrid(kC * kC), 256, 0, stream>>>(W_dw, Wdwb, kC * kC);

    // static GCN path
    dgc_k1_static_adj<<<kB * (kC / 16) * (kN / 16) / 8, 256, 0, stream>>>(xb, Wsab, tT);
    dgc_k2_static_weight<<<kB * (kC / 16) * (kN / 16) / 8, 256, 0, stream>>>(Wswb, tT, x, b_sw, x1b);

    // attention path
    dgc_k3_energy<<<kB * (kC / 16) * (kC / 16) / 8, 256, 0, stream>>>(xb, energ);
    dgc_k4a_rowmax<<<kB * kC, 64, 0, stream>>>(energ, en);
    dgc_k4b_bn_attn<<<kC, 256, 0, stream>>>(en, bn_g, bn_b, attn);
    dgc_k5_xglb<<<kB * (kC / 16) * (kN / 16) / 8, 256, 0, stream>>>(attn, xT, x, gamma, xgT);

    // fused dynamic GCN (dyn_adj stays in LDS)
    dgc_k6_dynamic<<<kB * (kN / 64), 512, 0, stream>>>(Wcmb, xgT, x1b, Wdwb, b_cm, b_dw, y);
}
